// SMPNNBlock_14731737825825
// MI455X (gfx1250) — matrix operand order repaired and checked by reading, not compiled, not run on hardware
//
#include <hip/hip_runtime.h>

typedef __bf16 bf16;
typedef __attribute__((ext_vector_type(16))) __bf16 v16bf;
typedef __attribute__((ext_vector_type(8)))  __bf16 v8bf;
typedef __attribute__((ext_vector_type(4)))  __bf16 v4bf;
typedef __attribute__((ext_vector_type(8)))  float  v8f;

#define DIM 128

__device__ __forceinline__ float wave_sum(float v) {
#pragma unroll
  for (int m = 16; m > 0; m >>= 1) v += __shfl_xor(v, m, 32);
  return v;
}

__device__ __forceinline__ float silu_f(float f) {
  return f / (1.0f + __expf(-f));
}

// ---- small weight transpose + fp32->bf16 convert: Wt[n][k] = W[k][n] ----
__global__ __launch_bounds__(256) void k_transpose_bf16(const float* __restrict__ W,
                                                        bf16* __restrict__ Wt) {
  int tid = blockIdx.x * 256 + threadIdx.x;   // 0..16383
  int n = tid >> 7, k = tid & 127;
  Wt[tid] = (bf16)W[(size_t)k * DIM + n];
}

// ---- degree pipeline ----
__global__ __launch_bounds__(256) void k_deg_init(float* __restrict__ deg, int n) {
  int t = blockIdx.x * 256 + threadIdx.x;
  if (t < n) deg[t] = 1.0f;   // self loop
}
__global__ __launch_bounds__(256) void k_deg_count(const long long* __restrict__ dst,
                                                   float* __restrict__ deg, int e) {
  int t = blockIdx.x * 256 + threadIdx.x;
  if (t < e) atomicAdd(&deg[(int)dst[t]], 1.0f);
}
__global__ __launch_bounds__(256) void k_deg_rsqrt(float* __restrict__ deg, int n) {
  int t = blockIdx.x * 256 + threadIdx.x;
  if (t < n) deg[t] = rsqrtf(deg[t]);
}

// ---- LayerNorm (one wave32 per 128-wide row) -> bf16 ----
__global__ __launch_bounds__(256) void k_layernorm_bf16(const float* __restrict__ x,
                                                        const float* __restrict__ g,
                                                        const float* __restrict__ b,
                                                        bf16* __restrict__ out, int nrows) {
  int row  = blockIdx.x * 8 + (threadIdx.x >> 5);
  int lane = threadIdx.x & 31;
  if (row >= nrows) return;
  const float4 v = *(const float4*)(x + (size_t)row * DIM + lane * 4);
  float s  = v.x + v.y + v.z + v.w;
  float sq = v.x * v.x + v.y * v.y + v.z * v.z + v.w * v.w;
  s = wave_sum(s); sq = wave_sum(sq);
  float mean = s * (1.0f / DIM);
  float inv  = rsqrtf(sq * (1.0f / DIM) - mean * mean + 1e-5f);
  float4 gg = *(const float4*)(g + lane * 4);
  float4 bb = *(const float4*)(b + lane * 4);
  v4bf o;
  o[0] = (bf16)((v.x - mean) * inv * gg.x + bb.x);
  o[1] = (bf16)((v.y - mean) * inv * gg.y + bb.y);
  o[2] = (bf16)((v.z - mean) * inv * gg.z + bb.z);
  o[3] = (bf16)((v.w - mean) * inv * gg.w + bb.w);
  *(v4bf*)(out + (size_t)row * DIM + lane * 4) = o;
}

// ---- cooperative 32KB weight stage into LDS (all 256 threads, before barrier) ----
__device__ __forceinline__ void stage_weights(const bf16* __restrict__ Bt, bf16* sB, int tid) {
  const uint4* src = (const uint4*)Bt;   // 2048 x 16B
  uint4* dst = (uint4*)sB;
#pragma unroll
  for (int i = 0; i < 8; ++i) {
    int idx = tid + i * 256;
    dst[idx] = src[idx];
  }
}

// ---- WMMA bf16 GEMM: C[N,128] = A[N,128] @ W, W transposed as Bt[n][k], B fed from LDS ----
__global__ __launch_bounds__(256) void k_gemm_bf16(const bf16* __restrict__ A,
                                                   const bf16* __restrict__ Bt,
                                                   float* __restrict__ C, int nrows) {
  __shared__ bf16 sB[DIM * DIM];                 // 32 KB of the 320 KB WGP pool
  stage_weights(Bt, sB, threadIdx.x);
  __syncthreads();

  int tile = blockIdx.x * 8 + (threadIdx.x >> 5);
  int lane = threadIdx.x & 31;
  int row0 = tile * 16;
  if (row0 >= nrows) return;                     // wave-uniform exit, after barrier
  int mr = lane & 15;                            // row in A tile / col in B tile
  int kg = lane >> 4;                            // k half: lane half kg holds k = 16*kg + s (contiguous)
  const bf16* arow = A + (size_t)(row0 + mr) * DIM;
  v16bf af[4];
#pragma unroll
  for (int kc = 0; kc < 4; ++kc) {
    v8bf lo = *(const v8bf*)(arow + kc * 32 + kg * 16);
    v8bf hi = *(const v8bf*)(arow + kc * 32 + kg * 16 + 8);
    af[kc] = __builtin_shufflevector(lo, hi, 0,1,2,3,4,5,6,7,8,9,10,11,12,13,14,15);
  }
  for (int nt = 0; nt < 8; ++nt) {
    v8f acc = {0.f,0.f,0.f,0.f,0.f,0.f,0.f,0.f};
    const bf16* bcol = sB + (nt * 16 + mr) * DIM + kg * 16;
#pragma unroll
    for (int kc = 0; kc < 4; ++kc) {
      v16bf bfrag = *(const v16bf*)(bcol + kc * 32);   // 2x ds_load_b128
      acc = __builtin_amdgcn_wmma_f32_16x16x32_bf16(false, af[kc], false, bfrag,
                                                    (short)0, acc, false, false);
    }
    float* cbase = C + (size_t)row0 * DIM + nt * 16 + mr;
#pragma unroll
    for (int r = 0; r < 8; ++r)
      cbase[(size_t)(r + kg * 8) * DIM] = acc[r];
  }
}

// ---- self-loop term: acc[i,:] = dinv[i]^2 * h[i,:]  (also zero-initializes acc) ----
__global__ __launch_bounds__(256) void k_selfloop(const float* __restrict__ h,
                                                  const float* __restrict__ dinv,
                                                  float* __restrict__ acc, int nrows) {
  int t = blockIdx.x * 256 + threadIdx.x;     // nrows*32 threads
  int row = t >> 5;
  if (row >= nrows) return;
  int c = (t & 31) * 4;
  float di = dinv[row];
  float w = di * di;
  float4 hv = *(const float4*)(h + (size_t)row * DIM + c);
  float4 o = {hv.x * w, hv.y * w, hv.z * w, hv.w * w};
  *(float4*)(acc + (size_t)row * DIM + c) = o;
}

// ---- edge scatter: one wave32 per edge, 4 floats/lane, f32 atomics resolve in L2 ----
__global__ __launch_bounds__(256) void k_scatter(const long long* __restrict__ ei,
                                                 const float* __restrict__ h,
                                                 const float* __restrict__ dinv,
                                                 float* __restrict__ acc, int e) {
  int edge = blockIdx.x * 8 + (threadIdx.x >> 5);
  int lane = threadIdx.x & 31;
  if (edge >= e) return;
  int s = (int)ei[edge];
  int d = (int)ei[(size_t)e + edge];
  float norm = dinv[s] * dinv[d];
  float4 hv = *(const float4*)(h + (size_t)s * DIM + lane * 4);
  float* ap = acc + (size_t)d * DIM + lane * 4;
  atomicAdd(ap + 0, hv.x * norm);
  atomicAdd(ap + 1, hv.y * norm);
  atomicAdd(ap + 2, hv.z * norm);
  atomicAdd(ap + 3, hv.w * norm);
}

// ---- fused: m=silu(acc+gcn_b); x1=x+a1*m; h2=LN2(x1)->bf16 ----
__global__ __launch_bounds__(256) void k_agg_ln2(const float* __restrict__ x,
                                                 const float* __restrict__ acc,
                                                 const float* __restrict__ gcnb,
                                                 const float* __restrict__ alpha1,
                                                 const float* __restrict__ g2,
                                                 const float* __restrict__ b2,
                                                 float* __restrict__ x1,
                                                 bf16* __restrict__ h2, int nrows) {
  int row  = blockIdx.x * 8 + (threadIdx.x >> 5);
  int lane = threadIdx.x & 31;
  if (row >= nrows) return;
  float a1 = *alpha1;
  float4 av = *(const float4*)(acc + (size_t)row * DIM + lane * 4);
  float4 gb = *(const float4*)(gcnb + lane * 4);
  float4 xv = *(const float4*)(x + (size_t)row * DIM + lane * 4);
  float4 x1v;
  x1v.x = xv.x + a1 * silu_f(av.x + gb.x);
  x1v.y = xv.y + a1 * silu_f(av.y + gb.y);
  x1v.z = xv.z + a1 * silu_f(av.z + gb.z);
  x1v.w = xv.w + a1 * silu_f(av.w + gb.w);
  *(float4*)(x1 + (size_t)row * DIM + lane * 4) = x1v;
  float s  = x1v.x + x1v.y + x1v.z + x1v.w;
  float sq = x1v.x * x1v.x + x1v.y * x1v.y + x1v.z * x1v.z + x1v.w * x1v.w;
  s = wave_sum(s); sq = wave_sum(sq);
  float mean = s * (1.0f / DIM);
  float inv  = rsqrtf(sq * (1.0f / DIM) - mean * mean + 1e-5f);
  float4 gg = *(const float4*)(g2 + lane * 4);
  float4 bb = *(const float4*)(b2 + lane * 4);
  v4bf o;
  o[0] = (bf16)((x1v.x - mean) * inv * gg.x + bb.x);
  o[1] = (bf16)((x1v.y - mean) * inv * gg.y + bb.y);
  o[2] = (bf16)((x1v.z - mean) * inv * gg.z + bb.z);
  o[3] = (bf16)((x1v.w - mean) * inv * gg.w + bb.w);
  *(v4bf*)(h2 + (size_t)row * DIM + lane * 4) = o;
}

// ---- GEMM2 with fused epilogue: out = x1 + a2*silu(h2@ff_w + ff_b), B from LDS ----
__global__ __launch_bounds__(256) void k_gemm2_epi(const bf16* __restrict__ A,
                                                   const bf16* __restrict__ Bt,
                                                   const float* __restrict__ ffb,
                                                   const float* __restrict__ x1,
                                                   const float* __restrict__ alpha2,
                                                   float* __restrict__ out, int nrows) {
  __shared__ bf16 sB[DIM * DIM];
  stage_weights(Bt, sB, threadIdx.x);
  __syncthreads();

  int tile = blockIdx.x * 8 + (threadIdx.x >> 5);
  int lane = threadIdx.x & 31;
  int row0 = tile * 16;
  if (row0 >= nrows) return;
  int mr = lane & 15;
  int kg = lane >> 4;
  float a2 = *alpha2;
  const bf16* arow = A + (size_t)(row0 + mr) * DIM;
  v16bf af[4];
#pragma unroll
  for (int kc = 0; kc < 4; ++kc) {
    v8bf lo = *(const v8bf*)(arow + kc * 32 + kg * 16);
    v8bf hi = *(const v8bf*)(arow + kc * 32 + kg * 16 + 8);
    af[kc] = __builtin_shufflevector(lo, hi, 0,1,2,3,4,5,6,7,8,9,10,11,12,13,14,15);
  }
  for (int nt = 0; nt < 8; ++nt) {
    v8f acc = {0.f,0.f,0.f,0.f,0.f,0.f,0.f,0.f};
    const bf16* bcol = sB + (nt * 16 + mr) * DIM + kg * 16;
#pragma unroll
    for (int kc = 0; kc < 4; ++kc) {
      v16bf bfrag = *(const v16bf*)(bcol + kc * 32);
      acc = __builtin_amdgcn_wmma_f32_16x16x32_bf16(false, af[kc], false, bfrag,
                                                    (short)0, acc, false, false);
    }
    int col = nt * 16 + mr;
    float bias = ffb[col];
#pragma unroll
    for (int r = 0; r < 8; ++r) {
      int m = r + kg * 8;
      size_t ix = (size_t)(row0 + m) * DIM + col;
      out[ix] = x1[ix] + a2 * silu_f(acc[r] + bias);
    }
  }
}

extern "C" void kernel_launch(void* const* d_in, const int* in_sizes, int n_in,
                              void* d_out, int out_size, void* d_ws, size_t ws_size,
                              hipStream_t stream) {
  (void)n_in; (void)out_size; (void)ws_size;
  const float*     x      = (const float*)d_in[0];
  const long long* ei     = (const long long*)d_in[1];   // int64 [2,E]
  const float*     ln1_g  = (const float*)d_in[2];
  const float*     ln1_b  = (const float*)d_in[3];
  const float*     gcn_w  = (const float*)d_in[4];
  const float*     gcn_b  = (const float*)d_in[5];
  const float*     alpha1 = (const float*)d_in[6];
  const float*     alpha2 = (const float*)d_in[7];
  const float*     ln2_g  = (const float*)d_in[8];
  const float*     ln2_b  = (const float*)d_in[9];
  const float*     ff_w   = (const float*)d_in[10];
  const float*     ff_b   = (const float*)d_in[11];

  const int n = in_sizes[0] / DIM;     // 200000
  const int e = in_sizes[1] / 2;       // 600000
  const size_t ND = (size_t)n * DIM;

  char* ws = (char*)d_ws;
  size_t off = 0;
  auto carve = [&](size_t bytes) { size_t o = off; off = (off + bytes + 255) & ~(size_t)255; return o; };
  float* deg = (float*)(ws + carve((size_t)n * 4));   // deg -> dinv (in place)
  bf16*  hbf = (bf16*) (ws + carve(ND * 2));          // h1 bf16, later reused for h2 bf16
  float* h   = (float*)(ws + carve(ND * 4));          // h = LN1(x)@W, later reused for x1
  float* acc = (float*)(ws + carve(ND * 4));          // scatter accumulator
  bf16*  wt  = (bf16*) (ws + carve((size_t)DIM * DIM * 2));
  bf16*  fft = (bf16*) (ws + carve((size_t)DIM * DIM * 2));

  // weights -> transposed bf16
  k_transpose_bf16<<<(DIM * DIM) / 256, 256, 0, stream>>>(gcn_w, wt);
  k_transpose_bf16<<<(DIM * DIM) / 256, 256, 0, stream>>>(ff_w, fft);

  // degrees
  k_deg_init <<<(n + 255) / 256, 256, 0, stream>>>(deg, n);
  k_deg_count<<<(e + 255) / 256, 256, 0, stream>>>(ei + (size_t)e, deg, e);
  k_deg_rsqrt<<<(n + 255) / 256, 256, 0, stream>>>(deg, n);

  // LN1 -> bf16
  k_layernorm_bf16<<<(n + 7) / 8, 256, 0, stream>>>(x, ln1_g, ln1_b, hbf, n);

  // h = h1 @ gcn_w   (WMMA bf16, weights staged in LDS)
  const int tiles = (n + 15) / 16;
  k_gemm_bf16<<<(tiles + 7) / 8, 256, 0, stream>>>(hbf, wt, h, n);

  // aggregation: self loops, then edge scatter-add
  k_selfloop<<<((size_t)n * 32 + 255) / 256, 256, 0, stream>>>(h, deg, acc, n);
  k_scatter <<<(e + 7) / 8, 256, 0, stream>>>(ei, h, deg, acc, e);

  // silu + residual + LN2 (x1 overwrites h buffer; h2 bf16 overwrites h1 buffer)
  k_agg_ln2<<<(n + 7) / 8, 256, 0, stream>>>(x, acc, gcn_b, alpha1, ln2_g, ln2_b, h, hbf, n);

  // out = x1 + a2*silu(h2 @ ff_w + ff_b)   (WMMA bf16, fused epilogue, weights in LDS)
  k_gemm2_epi<<<(tiles + 7) / 8, 256, 0, stream>>>(hbf, fft, ff_b, h, alpha2, (float*)d_out, n);
}
